// MultiScaleAttention_13580686590030
// MI455X (gfx1250) — compile-verified, matrix operand order checked by reading, not run on hardware
//
#include <hip/hip_runtime.h>

typedef __attribute__((ext_vector_type(16))) __bf16 v16bf;
typedef __attribute__((ext_vector_type(8)))  __bf16 v8bf;
typedef __attribute__((ext_vector_type(8)))  float  v8f;

#define NR_    216
#define TOKP   224      // padded tokens
#define HEADS_ 8
#define HD_    24
#define HDP    32       // padded head dim
#define CDIM   192
#define BD     2
#define DFULL  48
#define NTOK   (DFULL*DFULL*DFULL)
#define SCALE_Q 0.20412414523193154f   // 24^-0.5
#define DWT_F   0.35355339059327373f   // (1/sqrt(2))^3

// Fragment element e -> K offset (16x32 bf16 WMMA layout): e0..7 -> 8*hi.., e8..15 -> 16+8*hi..
// A fragment = two contiguous 8-element runs at p and p+16 (p already offset by 8*hi).
__device__ __forceinline__ v16bf frag_bf16(const __bf16* p) {
  v8bf lo = *(const v8bf*)p;
  v8bf hi = *(const v8bf*)(p + 16);
  return __builtin_shufflevector(lo, hi, 0,1,2,3,4,5,6,7,8,9,10,11,12,13,14,15);
}
__device__ __forceinline__ v16bf frag_f32(const float* p) {
  v16bf a;
#pragma unroll
  for (int e = 0; e < 8; ++e) {
    a[e]     = (__bf16)p[e];
    a[e + 8] = (__bf16)p[e + 16];
  }
  return a;
}

// ---------------- zero fill (16B granules) ----------------
__global__ void k_fill(uint4* __restrict__ p, long n) {
  long t = (long)blockIdx.x * blockDim.x + threadIdx.x;
  if (t < n) p[t] = make_uint4(0u, 0u, 0u, 0u);
}

// ---------------- biasT: [head][key 224][query 224], zero padded ----------------
__global__ void k_biasT(const float* __restrict__ bt, float* __restrict__ o) {
  int t = blockIdx.x * blockDim.x + threadIdx.x;
  if (t >= TOKP * TOKP) return;
  int key = t / TOKP, qu = t % TOKP;
  bool valid = (key < NR_) && (qu < NR_);
  int idx = 0;
  if (valid) {
    int qz = qu / 36, qy = (qu / 6) % 6, qx = qu % 6;
    int kz = key / 36, ky = (key / 6) % 6, kx = key % 6;
    idx = (qz - kz + 5) * 17 + (qy - ky + 5) * 11 + (qx - kx + 5);
  }
  for (int h = 0; h < HEADS_; ++h)
    o[((long)h * TOKP + key) * TOKP + qu] = valid ? bt[idx * HEADS_ + h] : 0.f;
}

// ---------------- weight transpose+convert: o[n*K+k] = (bf16)w[k*N+n] ----------------
__global__ void k_wt(const float* __restrict__ w, __bf16* __restrict__ o, int K, int N) {
  int t = blockIdx.x * blockDim.x + threadIdx.x;
  if (t >= K * N) return;
  int n = t / K, k = t % K;
  o[t] = (__bf16)w[(long)k * N + n];
}

// ---------------- haar DWT level-1 approx: 2x2x2 sum * 0.5^1.5 ----------------
// dual output: f32 (exact chain for next level) + bf16 (WMMA A operand)
__global__ void k_dwt(const float* __restrict__ in, float* __restrict__ out,
                      __bf16* __restrict__ outb, int dOut) {
  long total = (long)BD * dOut * dOut * dOut * CDIM;
  long t = (long)blockIdx.x * blockDim.x + threadIdx.x;
  if (t >= total) return;
  int c = (int)(t % CDIM); long r = t / CDIM;
  int x = (int)(r % dOut); r /= dOut;
  int y = (int)(r % dOut); r /= dOut;
  int z = (int)(r % dOut); int b = (int)(r / dOut);
  int dIn = dOut * 2;
  float s = 0.f;
  for (int dz = 0; dz < 2; ++dz)
    for (int dy = 0; dy < 2; ++dy)
      for (int dx = 0; dx < 2; ++dx) {
        long off = ((((long)b * dIn + (2 * z + dz)) * dIn + (2 * y + dy)) * dIn + (2 * x + dx)) * CDIM + c;
        s += in[off];
      }
  float v = s * DWT_F;
  out[t]  = v;
  outb[t] = (__bf16)v;
}

// ---------------- windowed QKV GEMM: per window M=216 N=576 K=192 ----------------
// qkv layout (padded, pre-zeroed): [j][win][head][224][32] bf16 ; q pre-scaled
__global__ void k_qkv(const __bf16* __restrict__ xlb, const __bf16* __restrict__ wqkvT,
                      __bf16* __restrict__ qkv, int ds, int nreg, int Bw) {
  int win  = blockIdx.x;
  int wave = threadIdx.x >> 5;
  int lane = threadIdx.x & 31;
  int row = lane & 15, hi = lane >> 4;
  int tile = blockIdx.y * 4 + wave;        // 0..503 (14 M-tiles x 36 N-tiles)
  int mt = tile / 36, nt = tile % 36;
  int mbase = mt * 16, nbase = nt * 16;

  int b = win / nreg, reg = win % nreg;
  int nrd = ds / 6;
  int rz = reg / (nrd * nrd), ry = (reg / nrd) % nrd, rx = reg % nrd;

  int mc = min(mbase + row, NR_ - 1);      // clamp: garbage rows are discarded on store
  int tz = mc / 36, ty = (mc / 6) % 6, tx = mc % 6;
  int gz = rz * 6 + tz, gy = ry * 6 + ty, gx = rx * 6 + tx;
  const __bf16* arow = xlb + ((((long)b * ds + gz) * ds + gy) * ds + gx) * CDIM + 8 * hi;
  const __bf16* brow = wqkvT + (long)(nbase + row) * CDIM + 8 * hi;

  v8f c = {};
  for (int kk = 0; kk < CDIM; kk += 32) {
    v16bf a  = frag_bf16(arow + kk);
    v16bf bb = frag_bf16(brow + kk);
    c = __builtin_amdgcn_wmma_f32_16x16x32_bf16(false, a, false, bb, (short)0, c, false, false);
  }
#pragma unroll
  for (int r = 0; r < 8; ++r) {
    int mo = mbase + r + 8 * hi;
    if (mo >= NR_) continue;
    int n = nbase + (lane & 15);
    int j = n / CDIM, rem = n % CDIM;
    int head = rem / HD_, dd = rem % HD_;
    float v = c[r];
    if (j == 0) v *= SCALE_Q;
    long base = (long)j * Bw * HEADS_ + (long)win * HEADS_ + head;
    qkv[(base * TOKP + mo) * HDP + dd] = (__bf16)v;
  }
}

// ---------------- attention per (window, head); 64 threads = 2 waves ----------------
__global__ void k_attn(const __bf16* __restrict__ qkv, const float* __restrict__ biasT,
                       float* __restrict__ y, int nreg, int Bw) {
  __shared__ float  S[2][16][TOKP];
  __shared__ __bf16 Vt[HDP][TOKP];
  int win  = blockIdx.x / HEADS_;
  int head = blockIdx.x % HEADS_;
  long plane = (long)Bw * HEADS_ * TOKP * HDP;
  const __bf16* qp = qkv + ((long)win * HEADS_ + head) * TOKP * HDP;
  const __bf16* kp = qp + plane;
  const __bf16* vp = kp + plane;
  int b = win / nreg, reg = win % nreg;
  float* yb = y + (long)b * CDIM * (nreg * NR_);

  // stage V^T into LDS (coalesced global read, LDS scatter)
  for (int idx = threadIdx.x; idx < TOKP * HDP; idx += 64) {
    int tok = idx >> 5, dd = idx & 31;
    Vt[dd][tok] = vp[idx];
  }
  __syncthreads();

  int wave = threadIdx.x >> 5, lane = threadIdx.x & 31;
  int row = lane & 15, hi = lane >> 4;

  for (int t = wave; t < 14; t += 2) {
    int mbase = t * 16;
    const __bf16* qrow = qp + (long)(mbase + row) * HDP + 8 * hi;
    v16bf a = frag_bf16(qrow);      // invariant across key tiles
    // phase 1: S = Q K^T + bias
    for (int nt = 0; nt < 14; ++nt) {
      int nbase = nt * 16;
      if (nt < 13)  // prefetch next K-tile row (global_prefetch_b8)
        __builtin_prefetch(kp + (long)(nbase + 16 + row) * HDP, 0, 1);
      v16bf bb = frag_bf16(kp + (long)(nbase + row) * HDP + 8 * hi);
      v8f c = {};
      c = __builtin_amdgcn_wmma_f32_16x16x32_bf16(false, a, false, bb, (short)0, c, false, false);
      int nn = nbase + (lane & 15);
      const float* pb = biasT + ((long)head * TOKP + nn) * TOKP + mbase + 8 * hi;
      float4 b0 = *(const float4*)pb;
      float4 b1 = *(const float4*)(pb + 4);
#pragma unroll
      for (int r = 0; r < 8; ++r) {
        float brv = (r < 4) ? ((const float*)&b0)[r] : ((const float*)&b1)[r - 4];
        S[wave][r + 8 * hi][nn] = (nn < NR_) ? (c[r] + brv) : -1e30f;
      }
    }
    // phase 2: row softmax (lanes 0..15, one row each; intra-wave LDS is in-order)
    if (lane < 16) {
      float4* Srow = (float4*)&S[wave][lane][0];
      float mx = -1e30f;
      for (int i = 0; i < NR_ / 4; ++i) {
        float4 v = Srow[i];
        mx = fmaxf(mx, fmaxf(fmaxf(v.x, v.y), fmaxf(v.z, v.w)));
      }
      float sum = 0.f;
      for (int i = 0; i < NR_ / 4; ++i) {
        float4 v = Srow[i];
        v.x = __expf(v.x - mx); v.y = __expf(v.y - mx);
        v.z = __expf(v.z - mx); v.w = __expf(v.w - mx);
        sum += v.x + v.y + v.z + v.w;
        Srow[i] = v;
      }
      float inv = 1.f / sum;
      for (int i = 0; i < NR_ / 4; ++i) {
        float4 v = Srow[i];
        v.x *= inv; v.y *= inv; v.z *= inv; v.w *= inv;
        Srow[i] = v;
      }
      Srow[54] = make_float4(0.f, 0.f, 0.f, 0.f);
      Srow[55] = make_float4(0.f, 0.f, 0.f, 0.f);
    }
    // phase 3: Y = P V (2 hd-tiles, K over 224 padded tokens)
    for (int dt = 0; dt < 2; ++dt) {
      int dd = dt * 16 + (lane & 15);
      v8f c = {};
      for (int kk = 0; kk < TOKP; kk += 32) {
        v16bf pa = frag_f32(&S[wave][row][kk + 8 * hi]);
        v16bf bb = frag_bf16(&Vt[dd][kk + 8 * hi]);
        c = __builtin_amdgcn_wmma_f32_16x16x32_bf16(false, pa, false, bb, (short)0, c, false, false);
      }
#pragma unroll
      for (int r = 0; r < 8; ++r) {
        int mm = mbase + r + 8 * hi;
        if (mm < NR_ && dd < HD_) {
          int ch = head * HD_ + dd;
          yb[(long)ch * (nreg * NR_) + reg * NR_ + mm] = c[r];  // quirk: raster = reg*216+tok
        }
      }
    }
  }
}

// ---------------- trilinear sample (half-pixel centers, edge clamp) ----------------
__device__ __forceinline__ float tri_sample(const float* base, int ss, float fz, float fy, float fx) {
  int z0 = (int)floorf(fz), y0 = (int)floorf(fy), x0 = (int)floorf(fx);
  float wz = fz - (float)z0, wy = fy - (float)y0, wx = fx - (float)x0;
  int z0c = min(max(z0, 0), ss - 1), z1c = min(max(z0 + 1, 0), ss - 1);
  int y0c = min(max(y0, 0), ss - 1), y1c = min(max(y0 + 1, 0), ss - 1);
  int x0c = min(max(x0, 0), ss - 1), x1c = min(max(x0 + 1, 0), ss - 1);
  float v000 = base[(z0c * ss + y0c) * ss + x0c], v001 = base[(z0c * ss + y0c) * ss + x1c];
  float v010 = base[(z0c * ss + y1c) * ss + x0c], v011 = base[(z0c * ss + y1c) * ss + x1c];
  float v100 = base[(z1c * ss + y0c) * ss + x0c], v101 = base[(z1c * ss + y0c) * ss + x1c];
  float v110 = base[(z1c * ss + y1c) * ss + x0c], v111 = base[(z1c * ss + y1c) * ss + x1c];
  float c00 = v000 + (v001 - v000) * wx, c01 = v010 + (v011 - v010) * wx;
  float c10 = v100 + (v101 - v100) * wx, c11 = v110 + (v111 - v110) * wx;
  float c0 = c00 + (c01 - c00) * wy, c1 = c10 + (c11 - c10) * wy;
  return c0 + (c1 - c0) * wz;
}

__device__ __forceinline__ float scale_sample(const float* yy, int ss, int b, int c,
                                              int z, int y, int x) {
  float sc = (float)ss * (1.f / 48.f);
  float fz = ((float)z + 0.5f) * sc - 0.5f;
  float fy = ((float)y + 0.5f) * sc - 0.5f;
  float fx = ((float)x + 0.5f) * sc - 0.5f;
  const float* base = yy + ((long)b * CDIM + c) * ss * ss * ss;
  return tri_sample(base, ss, fz, fy, fx);
}

// ---------------- upsample-accumulate + output projection ----------------
__global__ void k_final(const float* __restrict__ y1, const float* __restrict__ y2,
                        const float* __restrict__ y3, const __bf16* __restrict__ wprojT,
                        const float* __restrict__ bproj, float* __restrict__ out) {
  __shared__ __bf16 accS[32][CDIM];
  int tok0 = blockIdx.x * 32;
  int b  = tok0 / NTOK;
  int n0 = tok0 % NTOK;
  for (int idx = threadIdx.x; idx < 32 * CDIM; idx += blockDim.x) {
    int lt = idx / CDIM, c = idx % CDIM;
    int nn = n0 + lt;
    int z = nn / (DFULL * DFULL), yy = (nn / DFULL) % DFULL, x = nn % DFULL;
    float acc = scale_sample(y1, 24, b, c, z, yy, x)
              + scale_sample(y2, 12, b, c, z, yy, x)
              + scale_sample(y3,  6, b, c, z, yy, x);
    accS[lt][c] = (__bf16)acc;
  }
  __syncthreads();
  int wave = threadIdx.x >> 5, lane = threadIdx.x & 31;
  int row = lane & 15, hi = lane >> 4;
  for (int wt = wave; wt < 24; wt += 8) {
    int mt = wt / 12, nt = wt % 12;
    const __bf16* brow = wprojT + (long)(nt * 16 + row) * CDIM + 8 * hi;
    v8f c = {};
    for (int kk = 0; kk < CDIM; kk += 32) {
      v16bf a  = frag_bf16(&accS[mt * 16 + row][kk + 8 * hi]);
      v16bf bb = frag_bf16(brow + kk);
      c = __builtin_amdgcn_wmma_f32_16x16x32_bf16(false, a, false, bb, (short)0, c, false, false);
    }
#pragma unroll
    for (int r = 0; r < 8; ++r) {
      int lt = mt * 16 + r + 8 * hi;
      int n  = nt * 16 + (lane & 15);
      out[((long)b * NTOK + n0 + lt) * CDIM + n] = c[r] + bproj[n];
    }
  }
}

extern "C" void kernel_launch(void* const* d_in, const int* in_sizes, int n_in,
                              void* d_out, int out_size, void* d_ws, size_t ws_size,
                              hipStream_t stream) {
  (void)in_sizes; (void)n_in; (void)out_size; (void)ws_size;
  const float* x     = (const float*)d_in[0];
  const float* wqkv  = (const float*)d_in[1];
  const float* wproj = (const float*)d_in[2];
  const float* bproj = (const float*)d_in[3];
  const float* btab  = (const float*)d_in[4];
  float* out = (float*)d_out;

  char* ws = (char*)d_ws;
  size_t cur = 0;
  auto alloc = [&](size_t bytes) -> void* {
    void* p = ws + cur;
    cur += (bytes + 255) & ~(size_t)255;
    return p;
  };
  float*  biasT  = (float*) alloc((size_t)8 * TOKP * TOKP * 4);
  float*  xl1    = (float*) alloc((size_t)5308416 * 4);
  float*  xl2    = (float*) alloc((size_t)663552 * 4);
  float*  xl3    = (float*) alloc((size_t)82944 * 4);
  __bf16* xlb1   = (__bf16*)alloc((size_t)5308416 * 2);
  __bf16* xlb2   = (__bf16*)alloc((size_t)663552 * 2);
  __bf16* xlb3   = (__bf16*)alloc((size_t)82944 * 2);
  __bf16* wqkvT  = (__bf16*)alloc((size_t)576 * 192 * 2);
  __bf16* wprojT = (__bf16*)alloc((size_t)192 * 192 * 2);
  // padded qkv planes, allocated contiguously (sizes are multiples of 256B)
  const size_t qkv1e = (size_t)3 * 128 * HEADS_ * TOKP * HDP;  // 22,020,096
  const size_t qkv2e = (size_t)3 * 16  * HEADS_ * TOKP * HDP;  //  2,752,512
  const size_t qkv3e = (size_t)3 * 2   * HEADS_ * TOKP * HDP;  //    344,064
  __bf16* qkv1 = (__bf16*)alloc(qkv1e * 2);
  __bf16* qkv2 = (__bf16*)alloc(qkv2e * 2);
  __bf16* qkv3 = (__bf16*)alloc(qkv3e * 2);
  float*  y1   = (float*) alloc((size_t)5308416 * 4);
  float*  y2   = (float*) alloc((size_t)663552 * 4);
  float*  y3   = (float*) alloc((size_t)82944 * 4);

  // zero-fill all padded qkv storage (one contiguous region)
  long n4 = (long)((qkv1e + qkv2e + qkv3e) * 2 / 16);
  k_fill<<<(int)((n4 + 255) / 256), 256, 0, stream>>>((uint4*)qkv1, n4);

  k_biasT<<<(TOKP * TOKP + 255) / 256, 256, 0, stream>>>(btab, biasT);
  k_wt<<<(576 * 192 + 255) / 256, 256, 0, stream>>>(wqkv, wqkvT, 192, 576);
  k_wt<<<(192 * 192 + 255) / 256, 256, 0, stream>>>(wproj, wprojT, 192, 192);

  k_dwt<<<(5308416 + 255) / 256, 256, 0, stream>>>(x,   xl1, xlb1, 24);
  k_dwt<<<(663552 + 255) / 256, 256, 0, stream>>>(xl1, xl2, xlb2, 12);
  k_dwt<<<(82944 + 255) / 256, 256, 0, stream>>>(xl2, xl3, xlb3, 6);

  k_qkv<<<dim3(128, 126), 128, 0, stream>>>(xlb1, wqkvT, qkv1, 24, 64, 128);
  k_qkv<<<dim3(16, 126),  128, 0, stream>>>(xlb2, wqkvT, qkv2, 12, 8,  16);
  k_qkv<<<dim3(2, 126),   128, 0, stream>>>(xlb3, wqkvT, qkv3, 6,  1,  2);

  k_attn<<<128 * 8, 64, 0, stream>>>(qkv1, biasT, y1, 64, 128);
  k_attn<<<16 * 8,  64, 0, stream>>>(qkv2, biasT, y2, 8,  16);
  k_attn<<<2 * 8,   64, 0, stream>>>(qkv3, biasT, y3, 1,  2);

  k_final<<<6912, 256, 0, stream>>>(y1, y2, y3, wprojT, bproj, out);
}
